// LSTMModel_57080115363938
// MI455X (gfx1250) — matrix operand order repaired and checked by reading, not compiled, not run on hardware
//
#include <hip/hip_runtime.h>
#include <hip/hip_bf16.h>

typedef __bf16 bf16;
typedef __attribute__((ext_vector_type(16))) __bf16 v16bf;
typedef __attribute__((ext_vector_type(8)))  __bf16 v8bf;
typedef __attribute__((ext_vector_type(8)))  float  v8f;

#define BATCH 512
#define TSTEPS 120

__device__ __forceinline__ unsigned short f2bf(float f) {
    unsigned u = __float_as_uint(f);
    u += 0x7FFFu + ((u >> 16) & 1u);
    return (unsigned short)(u >> 16);
}
__device__ __forceinline__ float bf2f(unsigned short h) {
    return __uint_as_float(((unsigned)h) << 16);
}
__device__ __forceinline__ float sigmoidf_(float x) {
    return 1.0f / (1.0f + __expf(-x));
}

__device__ __forceinline__ v16bf load_frag(const bf16* p0, const bf16* p1) {
    v8bf lo = *(const v8bf*)p0;
    v8bf hi = *(const v8bf*)p1;
    v16bf f;
#pragma unroll
    for (int i = 0; i < 8; ++i) { f[i] = lo[i]; f[i + 8] = hi[i]; }
    return f;
}

// ---------------------------------------------------------------------------
// Pack combined [Wk ; Wr] (fp32, row-major [K,4U]) into BF16 WMMA B-fragment
// order. Element p = ((kt*NT + nt)*32 + lane)*16 + e holds
// W[kt*32 + (lane>>4)*16 + e, nt*16 + (lane&15)].
// ---------------------------------------------------------------------------
__global__ void pack_w_kernel(const float* __restrict__ Wk,
                              const float* __restrict__ Wr,
                              int Fin, int FinPad, int U, int Ktot,
                              unsigned short* __restrict__ out) {
    int total = Ktot * 4 * U;
    int p = blockIdx.x * blockDim.x + threadIdx.x;
    if (p >= total) return;
    int e    = p & 15;
    int lane = (p >> 4) & 31;
    int tile = p >> 9;
    int NT = (4 * U) >> 4;
    int nt = tile % NT;
    int kt = tile / NT;
    int k = kt * 32 + ((lane >> 4) * 16) + e;
    int n = nt * 16 + (lane & 15);
    float v = 0.0f;
    if (k < FinPad) {
        if (k < Fin) v = Wk[(long)k * (4 * U) + n];
    } else {
        v = Wr[(long)(k - FinPad) * (4 * U) + n];
    }
    out[p] = f2bf(v);
}

// Convert fp32 input x [512,120,132] -> bf16 padded [512,120,160]
__global__ void pack_x0_kernel(const float* __restrict__ x,
                               unsigned short* __restrict__ out) {
    int p = blockIdx.x * blockDim.x + threadIdx.x;
    int total = BATCH * TSTEPS * 160;
    if (p >= total) return;
    int kk = p % 160;
    int bt = p / 160;
    unsigned short v = 0;
    if (kk < 132) v = f2bf(x[(long)bt * 132 + kk]);
    out[p] = v;
}

__global__ void zero_u16_kernel(unsigned short* p, int n) {
    int i = blockIdx.x * blockDim.x + threadIdx.x;
    if (i < n) p[i] = 0;
}
__global__ void zero_f32_kernel(float* p, int n) {
    int i = blockIdx.x * blockDim.x + threadIdx.x;
    if (i < n) p[i] = 0.0f;
}

// ---------------------------------------------------------------------------
// One LSTM timestep, fully fused:
//   z = [x_t ; h_{t-1}] @ Wc + b   (BF16 WMMA, fp32 accumulate)
//   i,f,o = sigmoid; g = relu; c' = f*c + i*g; h' = o*relu(c')
// One wave owns a 32-row x 16-unit cell (2 M-tiles): each B fragment feeds
// two WMMAs, halving L2 weight traffic. The same wave computes all 4 gate
// tiles and does the elementwise state update. h double-buffered per step.
// ---------------------------------------------------------------------------
__global__ void __launch_bounds__(256)
lstm_step_kernel(const unsigned short* __restrict__ Xin, int finPad, int t,
                 const unsigned short* __restrict__ Wc,
                 const float* __restrict__ bias,
                 const unsigned short* __restrict__ hin,
                 unsigned short* __restrict__ hout,
                 float* __restrict__ cst,
                 unsigned short* __restrict__ Hseq,
                 int U) {
    const int lane = threadIdx.x & 31;
    const int wave = threadIdx.x >> 5;
    const int colTiles = U >> 4;
    const int cell = blockIdx.x * 8 + wave;
    const int rowPair = cell / colTiles;    // 0..15, 32 batch rows each
    const int colTile = cell % colTiles;
    const int NT = (4 * U) >> 4;

    const int aRow0 = rowPair * 32 + (lane & 15);
    const int aRow1 = aRow0 + 16;
    const int aSub = (lane >> 4) * 16;      // K sub-offset for this lane half

    v8f acc[8];                              // [rt*4 + g]
#pragma unroll
    for (int g = 0; g < 8; ++g)
#pragma unroll
        for (int i = 0; i < 8; ++i) acc[g][i] = 0.0f;

    const bf16* Xb = reinterpret_cast<const bf16*>(Xin);
    const bf16* Hb = reinterpret_cast<const bf16*>(hin);
    const bf16* Wb = reinterpret_cast<const bf16*>(Wc);

    // per-gate weight-fragment pointers (named scalars so the address-space
    // inference pass keeps them in the global aperture -> global_load_b128)
    const long wStride = (long)NT * 512;     // elements per kt step
    const long gStride = (long)colTiles * 512; // elements between gate tiles
    const bf16* wp0 = Wb + ((long)colTile * 32 + lane) * 16;
    const bf16* wp1 = wp0 + gStride;
    const bf16* wp2 = wp1 + gStride;
    const bf16* wp3 = wp2 + gStride;

    // ---- phase 1: A fragments from x_t ----
    {
        const bf16* ax0 = Xb + ((long)aRow0 * TSTEPS + t) * finPad + aSub;
        const bf16* ax1 = Xb + ((long)aRow1 * TSTEPS + t) * finPad + aSub;
        const int fKT = finPad >> 5;
        for (int kt = 0; kt < fKT; ++kt) {
            v16bf a0 = load_frag(ax0 + kt * 32, ax0 + kt * 32 + 8);
            v16bf a1 = load_frag(ax1 + kt * 32, ax1 + kt * 32 + 8);
            __builtin_prefetch(wp0 + wStride, 0, 1);
            v16bf b0 = load_frag(wp0, wp0 + 8);
            acc[0] = __builtin_amdgcn_wmma_f32_16x16x32_bf16(false, a0, false, b0, (short)0, acc[0], false, false);
            acc[4] = __builtin_amdgcn_wmma_f32_16x16x32_bf16(false, a1, false, b0, (short)0, acc[4], false, false);
            v16bf b1 = load_frag(wp1, wp1 + 8);
            acc[1] = __builtin_amdgcn_wmma_f32_16x16x32_bf16(false, a0, false, b1, (short)0, acc[1], false, false);
            acc[5] = __builtin_amdgcn_wmma_f32_16x16x32_bf16(false, a1, false, b1, (short)0, acc[5], false, false);
            v16bf b2 = load_frag(wp2, wp2 + 8);
            acc[2] = __builtin_amdgcn_wmma_f32_16x16x32_bf16(false, a0, false, b2, (short)0, acc[2], false, false);
            acc[6] = __builtin_amdgcn_wmma_f32_16x16x32_bf16(false, a1, false, b2, (short)0, acc[6], false, false);
            v16bf b3 = load_frag(wp3, wp3 + 8);
            acc[3] = __builtin_amdgcn_wmma_f32_16x16x32_bf16(false, a0, false, b3, (short)0, acc[3], false, false);
            acc[7] = __builtin_amdgcn_wmma_f32_16x16x32_bf16(false, a1, false, b3, (short)0, acc[7], false, false);
            wp0 += wStride; wp1 += wStride; wp2 += wStride; wp3 += wStride;
        }
    }
    // ---- phase 2: A fragments from h_{t-1} ----
    {
        const bf16* ah0 = Hb + (long)aRow0 * U + aSub;
        const bf16* ah1 = Hb + (long)aRow1 * U + aSub;
        const int hKT = U >> 5;
        for (int kt = 0; kt < hKT; ++kt) {
            v16bf a0 = load_frag(ah0 + kt * 32, ah0 + kt * 32 + 8);
            v16bf a1 = load_frag(ah1 + kt * 32, ah1 + kt * 32 + 8);
            __builtin_prefetch(wp0 + wStride, 0, 1);
            v16bf b0 = load_frag(wp0, wp0 + 8);
            acc[0] = __builtin_amdgcn_wmma_f32_16x16x32_bf16(false, a0, false, b0, (short)0, acc[0], false, false);
            acc[4] = __builtin_amdgcn_wmma_f32_16x16x32_bf16(false, a1, false, b0, (short)0, acc[4], false, false);
            v16bf b1 = load_frag(wp1, wp1 + 8);
            acc[1] = __builtin_amdgcn_wmma_f32_16x16x32_bf16(false, a0, false, b1, (short)0, acc[1], false, false);
            acc[5] = __builtin_amdgcn_wmma_f32_16x16x32_bf16(false, a1, false, b1, (short)0, acc[5], false, false);
            v16bf b2 = load_frag(wp2, wp2 + 8);
            acc[2] = __builtin_amdgcn_wmma_f32_16x16x32_bf16(false, a0, false, b2, (short)0, acc[2], false, false);
            acc[6] = __builtin_amdgcn_wmma_f32_16x16x32_bf16(false, a1, false, b2, (short)0, acc[6], false, false);
            v16bf b3 = load_frag(wp3, wp3 + 8);
            acc[3] = __builtin_amdgcn_wmma_f32_16x16x32_bf16(false, a0, false, b3, (short)0, acc[3], false, false);
            acc[7] = __builtin_amdgcn_wmma_f32_16x16x32_bf16(false, a1, false, b3, (short)0, acc[7], false, false);
            wp0 += wStride; wp1 += wStride; wp2 += wStride; wp3 += wStride;
        }
    }

    // ---- fused gate nonlinearities + state update ----
    const int u = colTile * 16 + (lane & 15);
    const float bi = bias[u];
    const float bff = bias[U + u];
    const float bg = bias[2 * U + u];
    const float bo = bias[3 * U + u];
#pragma unroll
    for (int rt = 0; rt < 2; ++rt) {
#pragma unroll
        for (int r = 0; r < 8; ++r) {
            // C/D layout: M = r + 8*(lane>>4)
            int b = rowPair * 32 + rt * 16 + ((lane >> 4) * 8) + r;
            float zi = acc[rt * 4 + 0][r] + bi;
            float zf = acc[rt * 4 + 1][r] + bff;
            float zg = acc[rt * 4 + 2][r] + bg;
            float zo = acc[rt * 4 + 3][r] + bo;
            long ci = (long)b * U + u;
            float cold = cst[ci];
            float cn = sigmoidf_(zf) * cold + sigmoidf_(zi) * fmaxf(zg, 0.0f);
            float hn = sigmoidf_(zo) * fmaxf(cn, 0.0f);
            cst[ci] = cn;
            unsigned short hb = f2bf(hn);
            hout[ci] = hb;
            Hseq[((long)b * TSTEPS + t) * U + u] = hb;
        }
    }
}

// ---------------------------------------------------------------------------
// Dense: out = act(A_bf16 @ W_f32 + b). Tiny fraction of FLOPs -> VALU.
// ---------------------------------------------------------------------------
__global__ void dense_kernel(const unsigned short* __restrict__ A, int aRowStride,
                             long aBase, const float* __restrict__ W,
                             const float* __restrict__ bias, int Fin, int Fout,
                             int doRelu, unsigned short* __restrict__ outBf,
                             float* __restrict__ outF) {
    int idx = blockIdx.x * blockDim.x + threadIdx.x;
    int total = BATCH * Fout;
    if (idx >= total) return;
    int b = idx / Fout, j = idx % Fout;
    const unsigned short* arow = A + aBase + (long)b * aRowStride;
    float s = bias[j];
    for (int k = 0; k < Fin; ++k) s += bf2f(arow[k]) * W[(long)k * Fout + j];
    if (doRelu) s = fmaxf(s, 0.0f);
    if (outBf) outBf[idx] = f2bf(s);
    else       outF[idx] = s;
}

__global__ void softmax_kernel(const float* __restrict__ logits,
                               float* __restrict__ out, int n) {
    __shared__ float red[128];
    int b = blockIdx.x, tid = threadIdx.x;
    float m = -1e30f;
    for (int j = tid; j < n; j += 128) m = fmaxf(m, logits[b * n + j]);
    red[tid] = m; __syncthreads();
    for (int s = 64; s > 0; s >>= 1) { if (tid < s) red[tid] = fmaxf(red[tid], red[tid + s]); __syncthreads(); }
    m = red[0]; __syncthreads();
    float sum = 0.0f;
    for (int j = tid; j < n; j += 128) sum += __expf(logits[b * n + j] - m);
    red[tid] = sum; __syncthreads();
    for (int s = 64; s > 0; s >>= 1) { if (tid < s) red[tid] += red[tid + s]; __syncthreads(); }
    float inv = 1.0f / red[0];
    for (int j = tid; j < n; j += 128) out[b * n + j] = __expf(logits[b * n + j] - m) * inv;
}

// ---------------------------------------------------------------------------
extern "C" void kernel_launch(void* const* d_in, const int* in_sizes, int n_in,
                              void* d_out, int out_size, void* d_ws, size_t ws_size,
                              hipStream_t stream) {
    (void)in_sizes; (void)n_in; (void)out_size; (void)ws_size;
    const float* x = (const float*)d_in[0];

    struct LCfg { int Fin, FinPad, U; };
    const LCfg L[5] = {{132,160,256},{256,256,256},{256,256,512},{512,512,512},{512,512,256}};

    char* ws = (char*)d_ws;
    size_t off = 0;
    auto take = [&](size_t bytes) -> void* {
        off = (off + 255) & ~(size_t)255;
        void* p = ws + off; off += bytes; return p;
    };

    unsigned short* xb0 = (unsigned short*)take((size_t)BATCH * TSTEPS * 512 * 2);
    unsigned short* xb1 = (unsigned short*)take((size_t)BATCH * TSTEPS * 512 * 2);
    unsigned short* wc[5];
    for (int l = 0; l < 5; ++l) {
        int Ktot = L[l].FinPad + L[l].U;
        wc[l] = (unsigned short*)take((size_t)Ktot * 4 * L[l].U * 2);
    }
    unsigned short* hb0 = (unsigned short*)take((size_t)BATCH * 512 * 2);
    unsigned short* hb1 = (unsigned short*)take((size_t)BATCH * 512 * 2);
    float* cst          = (float*)take((size_t)BATCH * 512 * 4);
    unsigned short* dt0 = (unsigned short*)take((size_t)BATCH * 256 * 2);
    unsigned short* dt1 = (unsigned short*)take((size_t)BATCH * 128 * 2);
    unsigned short* dt2 = (unsigned short*)take((size_t)BATCH * 128 * 2);
    float* logits       = (float*)take((size_t)BATCH * 101 * 4);

    // --- weight packing (once per call) ---
    for (int l = 0; l < 5; ++l) {
        int Ktot = L[l].FinPad + L[l].U;
        int total = Ktot * 4 * L[l].U;
        pack_w_kernel<<<(total + 255) / 256, 256, 0, stream>>>(
            (const float*)d_in[1 + 3 * l], (const float*)d_in[2 + 3 * l],
            L[l].Fin, L[l].FinPad, L[l].U, Ktot, wc[l]);
    }
    {
        int total = BATCH * TSTEPS * 160;
        pack_x0_kernel<<<(total + 255) / 256, 256, 0, stream>>>(x, xb0);
    }

    // --- LSTM stack ---
    unsigned short* xin = xb0;
    unsigned short* xout = xb1;
    for (int l = 0; l < 5; ++l) {
        const int U = L[l].U;
        const float* bias = (const float*)d_in[3 + 3 * l];
        int nState = BATCH * U;
        zero_u16_kernel<<<(nState + 255) / 256, 256, 0, stream>>>(hb0, nState);
        zero_f32_kernel<<<(nState + 255) / 256, 256, 0, stream>>>(cst, nState);

        unsigned short* hin = hb0;
        unsigned short* hout = hb1;
        const int cells = (BATCH / 32) * (U / 16);   // 32-row cells
        const int blocks = cells / 8;
        for (int t = 0; t < TSTEPS; ++t) {
            lstm_step_kernel<<<blocks, 256, 0, stream>>>(
                xin, L[l].FinPad, t, wc[l], bias, hin, hout, cst, xout, U);
            unsigned short* tmp = hin; hin = hout; hout = tmp;
        }
        unsigned short* tx = xin; xin = xout; xout = tx;
    }

    // --- dense head on h at t = T-1 (xin = layer-4 sequence, width 256) ---
    const float* dW0 = (const float*)d_in[16]; const float* db0 = (const float*)d_in[17];
    const float* dW1 = (const float*)d_in[18]; const float* db1 = (const float*)d_in[19];
    const float* dW2 = (const float*)d_in[20]; const float* db2 = (const float*)d_in[21];
    const float* dW3 = (const float*)d_in[22]; const float* db3 = (const float*)d_in[23];

    dense_kernel<<<(BATCH * 256 + 255) / 256, 256, 0, stream>>>(
        xin, TSTEPS * 256, (long)(TSTEPS - 1) * 256, dW0, db0, 256, 256, 1, dt0, nullptr);
    dense_kernel<<<(BATCH * 128 + 255) / 256, 256, 0, stream>>>(
        dt0, 256, 0, dW1, db1, 256, 128, 1, dt1, nullptr);
    dense_kernel<<<(BATCH * 128 + 255) / 256, 256, 0, stream>>>(
        dt1, 128, 0, dW2, db2, 128, 128, 1, dt2, nullptr);
    dense_kernel<<<(BATCH * 101 + 255) / 256, 256, 0, stream>>>(
        dt2, 128, 0, dW3, db3, 128, 101, 0, nullptr, logits);
    softmax_kernel<<<BATCH, 128, 0, stream>>>(logits, (float*)d_out, 101);
}
